// Attention_515396075526
// MI455X (gfx1250) — matrix operand order repaired and checked by reading, not compiled, not run on hardware
//
#include <hip/hip_runtime.h>
#include <hip/hip_bf16.h>

// ---- problem constants (match reference) ----
#define B_   2
#define L_   2048
#define DIM_ 4096
#define NH   32
#define NKV  8
#define HD   128
#define BL   (B_ * L_)            // 4096 rows
#define KVD  (NKV * HD)           // 1024

typedef __bf16 bf16;
typedef __attribute__((ext_vector_type(16))) __bf16 v16bf;
typedef __attribute__((ext_vector_type(8)))  float  v8f;
typedef __attribute__((ext_vector_type(4)))  __bf16 bf16x4;

union Frag {
    v16bf v;
    unsigned int u[8];
};

// ---------------------------------------------------------------------------
// elementwise f32 -> bf16
// ---------------------------------------------------------------------------
__global__ void cvt_f32_bf16(const float* __restrict__ in, bf16* __restrict__ out, int n) {
    int i = blockIdx.x * blockDim.x + threadIdx.x;
    int stride = gridDim.x * blockDim.x;
    for (; i < n; i += stride) out[i] = (bf16)in[i];
}

// ---------------------------------------------------------------------------
// traditional (interleaved) RoPE + f32 -> bf16.  layout: [B][L][heads][HD]
// ---------------------------------------------------------------------------
__global__ void rope_cvt(const float* __restrict__ in, bf16* __restrict__ out, int heads) {
    int i = blockIdx.x * blockDim.x + threadIdx.x;
    int total = BL * heads * (HD / 2);
    int stride = gridDim.x * blockDim.x;
    for (; i < total; i += stride) {
        int p  = i % (HD / 2);
        int h  = (i / (HD / 2)) % heads;
        int bl = i / ((HD / 2) * heads);
        int l  = bl % L_;
        float freq = __powf(10000.0f, -(float)p * (1.0f / 64.0f)); // theta^(-2p/128)
        float ang  = (float)l * freq;
        float s, c;
        __sincosf(ang, &s, &c);
        size_t base = ((size_t)bl * heads + h) * HD + 2 * p;
        float x1 = in[base], x2 = in[base + 1];
        out[base]     = (bf16)(x1 * c - x2 * s);
        out[base + 1] = (bf16)(x1 * s + x2 * c);
    }
}

// ---------------------------------------------------------------------------
// C[M][N] = A[M][K] * W[N][K]^T   (bf16 in, f32 out) via v_wmma_f32_16x16x32_bf16
// block = 256 threads (8 waves in 2x4), macro tile 64(M) x 128(N),
// each wave owns a 32x32 tile = 2x2 WMMA accumulators.
// Tile staging uses CDNA5 async copies: global_load_async_to_lds_b128
// (no VGPR round-trip; ordered with s_wait_asynccnt + workgroup barrier).
// ---------------------------------------------------------------------------
__global__ __launch_bounds__(256) void gemm_bf16(const bf16* __restrict__ A,
                                                 const bf16* __restrict__ W,
                                                 float* __restrict__ C,
                                                 int M, int N, int K) {
    // +8 element row pad: rows are 80 B apart -> every 16-B async B128 store
    // into LDS stays 16-B aligned, per-lane 32-bit gathers stay 4-B aligned.
    __shared__ bf16 lA[64][40];
    __shared__ bf16 lB[128][40];

    const int tid  = threadIdx.x;
    const int wid  = tid >> 5;
    const int lane = tid & 31;
    const int lm   = lane & 15;
    const int hh   = lane >> 4;

    const int m0 = blockIdx.y * 64;
    const int n0 = blockIdx.x * 128;
    const int mW = (wid >> 2) * 32;   // 0 or 32
    const int nW = (wid & 3) * 32;    // 0,32,64,96

    v8f acc[2][2];
#pragma unroll
    for (int mi = 0; mi < 2; mi++)
#pragma unroll
        for (int ni = 0; ni < 2; ni++)
#pragma unroll
            for (int j = 0; j < 8; j++) acc[mi][ni][j] = 0.0f;

    // cooperative staging indices (16 B of A, 32 B of B per thread)
    const int ar = tid >> 2;           // A: 8 elems/thread (64x32)
    const int ac = (tid & 3) * 8;
    const int br = tid >> 1;           // B: 16 elems/thread (128x32)
    const int bc = (tid & 1) * 16;

    const unsigned ldsA  = (unsigned)(size_t)&lA[ar][ac];
    const unsigned ldsB0 = (unsigned)(size_t)&lB[br][bc];
    const unsigned ldsB1 = (unsigned)(size_t)&lB[br][bc + 8];

    for (int k0 = 0; k0 < K; k0 += 32) {
        __syncthreads();   // previous tile fully consumed before overwrite
        {
            const bf16* ga = &A[(size_t)(m0 + ar) * K + k0 + ac];
            const bf16* gb = &W[(size_t)(n0 + br) * K + k0 + bc];
            unsigned long long aAddr = (unsigned long long)ga;
            unsigned long long bAddr = (unsigned long long)gb;
            asm volatile("global_load_async_to_lds_b128 %0, %1, off"
                         :: "v"(ldsA), "v"(aAddr) : "memory");
            asm volatile("global_load_async_to_lds_b128 %0, %1, off"
                         :: "v"(ldsB0), "v"(bAddr) : "memory");
            asm volatile("global_load_async_to_lds_b128 %0, %1, off offset:16"
                         :: "v"(ldsB1), "v"(bAddr) : "memory");
            if (k0 + 32 < K) {
                __builtin_prefetch(ga + 32, 0, 3);
                __builtin_prefetch(gb + 32, 0, 3);
            }
        }
        asm volatile("s_wait_asynccnt 0x0" ::: "memory");
        __syncthreads();   // all waves' async tile data visible in LDS

        Frag a[2], b[2];
#pragma unroll
        for (int mi = 0; mi < 2; mi++) {
            const int row = mW + mi * 16 + lm;
#pragma unroll
            for (int j = 0; j < 4; j++) {
                a[mi].u[j]     = *(const unsigned int*)&lA[row][8 * hh + 2 * j];
                a[mi].u[4 + j] = *(const unsigned int*)&lA[row][16 + 8 * hh + 2 * j];
            }
        }
#pragma unroll
        for (int ni = 0; ni < 2; ni++) {
            const int row = nW + ni * 16 + lm;
#pragma unroll
            for (int j = 0; j < 8; j++)
                b[ni].u[j] = *(const unsigned int*)&lB[row][8 * hh + 2 * j + 8 * (j >> 2)];
        }
#pragma unroll
        for (int mi = 0; mi < 2; mi++)
#pragma unroll
            for (int ni = 0; ni < 2; ni++)
                acc[mi][ni] = __builtin_amdgcn_wmma_f32_16x16x32_bf16(
                    false, a[mi].v, false, b[ni].v, (short)0, acc[mi][ni], false, false);
    }

#pragma unroll
    for (int mi = 0; mi < 2; mi++)
#pragma unroll
        for (int ni = 0; ni < 2; ni++)
#pragma unroll
            for (int j = 0; j < 8; j++) {
                int gm = m0 + mW + mi * 16 + j + 8 * hh;
                int gn = n0 + nW + ni * 16 + lm;
                C[(size_t)gm * N + gn] = acc[mi][ni][j];
            }
}

// ---------------------------------------------------------------------------
// flash attention: 4 waves / block, each wave = one 16-row Q block.
// Q:[B][L][NH][HD] bf16 (roped), K:[B][L][NKV][HD] bf16 (roped),
// V:[B][L][NKV][HD] bf16, O:[B][L][NH][HD] bf16
// ---------------------------------------------------------------------------
__global__ __launch_bounds__(128) void attn_fwd(const bf16* __restrict__ Q,
                                                const bf16* __restrict__ Kt,
                                                const bf16* __restrict__ V,
                                                bf16* __restrict__ O) {
    __shared__ bf16 Vt[HD][36];       // V^T tile: [d][kpos], shared by all 4 waves
    __shared__ bf16 lP[4][16][36];    // per-wave P staging (C-layout -> A-layout)

    const int tid  = threadIdx.x;
    const int wid  = tid >> 5;
    const int lane = tid & 31;
    const int lm   = lane & 15;
    const int hh   = lane >> 4;

    const int b    = blockIdx.z;
    const int h    = blockIdx.y;
    const int kvh  = h >> 2;                 // REPEATS = 4
    const int q0wg = blockIdx.x * 64;
    const int q0   = q0wg + wid * 16;
    const float scale = 0.08838834764831845f; // 128^-0.5

    // --- load Q fragments once (16 x 128 = 4 bf16 A-frags) ---
    Frag qa[4];
    {
        const bf16* qb = Q + ((size_t)(b * L_ + q0 + lm) * NH + h) * HD;
#pragma unroll
        for (int kk = 0; kk < 4; kk++) {
#pragma unroll
            for (int j = 0; j < 4; j++) {
                qa[kk].u[j]     = *(const unsigned int*)(qb + 32 * kk + 8 * hh + 2 * j);
                qa[kk].u[4 + j] = *(const unsigned int*)(qb + 32 * kk + 16 + 8 * hh + 2 * j);
            }
        }
    }

    v8f o8[8];
#pragma unroll
    for (int t = 0; t < 8; t++)
#pragma unroll
        for (int j = 0; j < 8; j++) o8[t][j] = 0.0f;

    float mrow[8], lrow[8];
#pragma unroll
    for (int j = 0; j < 8; j++) { mrow[j] = -3.0e38f; lrow[j] = 0.0f; }

    const int nkt = (q0wg + 64) / 32;   // causal: only k-tiles up to WG's max q row

    for (int kt = 0; kt < nkt; kt++) {
        const int k0 = kt * 32;

        // --- stage V^T tile cooperatively (transpose -> ds stores) ---
        __syncthreads();
        {
            int kk  = tid >> 2;            // 0..31 kpos in tile
            int dsg = (tid & 3) * 32;      // 32-wide d segment
            const bf16* vsrc = V + ((size_t)(b * L_ + k0 + kk) * NKV + kvh) * HD + dsg;
#pragma unroll
            for (int dd = 0; dd < 32; dd++) Vt[dsg + dd][kk] = vsrc[dd];
        }
        __syncthreads();

        // --- S = Q K^T over d=128 (two 16-col halves) ---
        v8f s[2];
#pragma unroll
        for (int ch = 0; ch < 2; ch++) {
            v8f sc;
#pragma unroll
            for (int j = 0; j < 8; j++) sc[j] = 0.0f;
            const int kp = k0 + ch * 16 + lm;
            const bf16* kb = Kt + ((size_t)(b * L_ + kp) * NKV + kvh) * HD;
#pragma unroll
            for (int kk = 0; kk < 4; kk++) {
                Frag bfr;
#pragma unroll
                for (int j = 0; j < 8; j++)
                    bfr.u[j] = *(const unsigned int*)(kb + 32 * kk + 8 * hh + 2 * j + 8 * (j >> 2));
                sc = __builtin_amdgcn_wmma_f32_16x16x32_bf16(false, qa[kk].v, false, bfr.v,
                                                             (short)0, sc, false, false);
            }
            s[ch] = sc;
        }

        // --- causal mask + online softmax (rows live on (j, lane-half)) ---
#pragma unroll
        for (int j = 0; j < 8; j++) {
            int qp  = q0 + j + 8 * hh;
            int kp0 = k0 + lm;
            int kp1 = k0 + 16 + lm;
            float s0 = s[0][j] * scale + ((kp0 > qp) ? -1.0e9f : 0.0f);
            float s1 = s[1][j] * scale + ((kp1 > qp) ? -1.0e9f : 0.0f);
            float v = fmaxf(s0, s1);
#pragma unroll
            for (int off = 1; off < 16; off <<= 1) v = fmaxf(v, __shfl_xor(v, off, 16));
            float mn    = fmaxf(mrow[j], v);
            float alpha = __expf(mrow[j] - mn);
            float p0    = __expf(s0 - mn);
            float p1    = __expf(s1 - mn);
            float rs    = p0 + p1;
#pragma unroll
            for (int off = 1; off < 16; off <<= 1) rs += __shfl_xor(rs, off, 16);
            lrow[j] = lrow[j] * alpha + rs;
            mrow[j] = mn;
#pragma unroll
            for (int t = 0; t < 8; t++) o8[t][j] = o8[t][j] * alpha;
            lP[wid][j + 8 * hh][lm]      = (bf16)p0;
            lP[wid][j + 8 * hh][16 + lm] = (bf16)p1;
        }
        asm volatile("s_wait_dscnt 0x0" ::: "memory"); // wave-local LDS RAW ordering

        // --- regather P as a 16x32 bf16 A-fragment ---
        Frag pa;
#pragma unroll
        for (int j = 0; j < 4; j++) {
            pa.u[j]     = *(const unsigned int*)&lP[wid][lm][8 * hh + 2 * j];
            pa.u[4 + j] = *(const unsigned int*)&lP[wid][lm][16 + 8 * hh + 2 * j];
        }

        // --- O += P V  (8 d-tiles of 16) ---
#pragma unroll
        for (int dt = 0; dt < 8; dt++) {
            Frag vfr;
#pragma unroll
            for (int j = 0; j < 8; j++)
                vfr.u[j] = *(const unsigned int*)&Vt[dt * 16 + lm][8 * hh + 2 * j + 8 * (j >> 2)];
            o8[dt] = __builtin_amdgcn_wmma_f32_16x16x32_bf16(false, pa.v, false, vfr.v,
                                                             (short)0, o8[dt], false, false);
        }
    }

    // --- normalize + write bf16 output [b][q][h][d] ---
#pragma unroll
    for (int j = 0; j < 8; j++) {
        int qp = q0 + j + 8 * hh;
        float inv = 1.0f / lrow[j];
#pragma unroll
        for (int dt = 0; dt < 8; dt++) {
            int d = dt * 16 + lm;
            O[((size_t)(b * L_ + qp) * NH + h) * HD + d] = (bf16)(o8[dt][j] * inv);
        }
    }
}

// ---------------------------------------------------------------------------
// host-side launcher
// ---------------------------------------------------------------------------
extern "C" void kernel_launch(void* const* d_in, const int* in_sizes, int n_in,
                              void* d_out, int out_size, void* d_ws, size_t ws_size,
                              hipStream_t stream) {
    (void)in_sizes; (void)n_in; (void)out_size; (void)ws_size;
    const float* x  = (const float*)d_in[0];
    // d_in[1] = mask (upper-tri -1e9): causal mask is computed inline
    const float* wq = (const float*)d_in[2];
    const float* wk = (const float*)d_in[3];
    const float* wv = (const float*)d_in[4];
    const float* wo = (const float*)d_in[5];
    float* out = (float*)d_out;

    char* p = (char*)d_ws;
    auto alloc = [&](size_t bytes) -> char* {
        char* r = p;
        p += (bytes + 255) & ~(size_t)255;
        return r;
    };
    bf16* xb  = (bf16*)alloc((size_t)BL * DIM_ * 2);
    bf16* wqb = (bf16*)alloc((size_t)DIM_ * DIM_ * 2);
    bf16* wkb = (bf16*)alloc((size_t)KVD * DIM_ * 2);
    bf16* wvb = (bf16*)alloc((size_t)KVD * DIM_ * 2);
    bf16* wob = (bf16*)alloc((size_t)DIM_ * DIM_ * 2);
    float* qf = (float*)alloc((size_t)BL * DIM_ * 4);
    float* kf = (float*)alloc((size_t)BL * KVD * 4);
    float* vf = (float*)alloc((size_t)BL * KVD * 4);
    bf16* qr  = (bf16*)alloc((size_t)BL * DIM_ * 2);
    bf16* kr  = (bf16*)alloc((size_t)BL * KVD * 2);
    bf16* vb  = (bf16*)alloc((size_t)BL * KVD * 2);
    bf16* ab  = (bf16*)alloc((size_t)BL * DIM_ * 2);

    const int T = 256;
    auto blk = [](int n, int t) { return (n + t - 1) / t; };

    // 1) convert activations + weights to bf16
    cvt_f32_bf16<<<blk(BL * DIM_, T),   T, 0, stream>>>(x,  xb,  BL * DIM_);
    cvt_f32_bf16<<<blk(DIM_ * DIM_, T), T, 0, stream>>>(wq, wqb, DIM_ * DIM_);
    cvt_f32_bf16<<<blk(KVD * DIM_, T),  T, 0, stream>>>(wk, wkb, KVD * DIM_);
    cvt_f32_bf16<<<blk(KVD * DIM_, T),  T, 0, stream>>>(wv, wvb, KVD * DIM_);
    cvt_f32_bf16<<<blk(DIM_ * DIM_, T), T, 0, stream>>>(wo, wob, DIM_ * DIM_);

    // 2) QKV projections (WMMA GEMMs): macro tile 64(M) x 128(N)
    gemm_bf16<<<dim3(DIM_ / 128, BL / 64), 256, 0, stream>>>(xb, wqb, qf, BL, DIM_, DIM_);
    gemm_bf16<<<dim3(KVD  / 128, BL / 64), 256, 0, stream>>>(xb, wkb, kf, BL, KVD,  DIM_);
    gemm_bf16<<<dim3(KVD  / 128, BL / 64), 256, 0, stream>>>(xb, wvb, vf, BL, KVD,  DIM_);

    // 3) RoPE (q, k) + convert v to bf16
    rope_cvt<<<blk(BL * NH  * (HD / 2), T), T, 0, stream>>>(qf, qr, NH);
    rope_cvt<<<blk(BL * NKV * (HD / 2), T), T, 0, stream>>>(kf, kr, NKV);
    cvt_f32_bf16<<<blk(BL * KVD, T), T, 0, stream>>>(vf, vb, BL * KVD);

    // 4) flash attention
    attn_fwd<<<dim3(L_ / 64, NH, B_), 128, 0, stream>>>(qr, kr, vb, ab);

    // 5) output projection -> f32 d_out
    gemm_bf16<<<dim3(DIM_ / 128, BL / 64), 256, 0, stream>>>(ab, wob, out, BL, DIM_, DIM_);
}
